// TuckER_7421703488185
// MI455X (gfx1250) — matrix operand order repaired and checked by reading, not compiled, not run on hardware
//
#include <hip/hip_runtime.h>
#include <hip/hip_bf16.h>

// TuckER forward on MI455X (gfx1250), bf16 WMMA + f32 accumulation.
//
// grad_mask is algebraically irrelevant for the forward value. Pipeline:
//   prep_s       : gather subject rows -> s_bf [1024][224] bf16 (zero-padded K)
//   zero_bf      : zero tmp2b pad region
//   prep_ent     : entity_emb f32 -> ent_bf [100000][224] bf16 (zero-padded K)
//   prep_coreT   : core[e,r,o] f32 -> coreT[r][o:208][e:224] bf16 (transposed, padded)
//   gemm1_contract: tmp2[b,o] = sum_r r[b,r] * (s[b,:] x core[:,r,o])  (WMMA + VALU fuse)
//   gemm2_sigmoid : out = sigmoid(tmp2 @ E^T)                          (WMMA, fused epilogue)

typedef __attribute__((ext_vector_type(16))) __bf16 v16bf;
typedef __attribute__((ext_vector_type(8)))  float  v8f;

#define DE  200      // entity dim
#define DR  200      // relation dim
#define NE  100000   // num entities
#define BATCH 1024
#define KP  224      // K padded to multiple of 32 (7 WMMA k-steps)
#define OPAD 208     // o padded to multiple of 16 (13 o-tiles)

union FragU { v16bf v; uint4 u[2]; };

// A fragment (16-bit A 16x32): lane l (row = l&15, khalf = l>>4) holds
// K = khalf*16 .. +15 in elems [0..15] (matches B fragment order).
// p = &Arow[kbase + khalf*16]; second run at p+8.
static __device__ __forceinline__ v16bf load_a_frag(const __bf16* p) {
  FragU f;
  f.u[0] = *reinterpret_cast<const uint4*>(p);
  f.u[1] = *reinterpret_cast<const uint4*>(p + 8);
  return f.v;
}

// B fragment (16-bit B 32x16): lane l (col = l&15, khalf = l>>4) holds
// K = khalf*16 .. +15 of its column, contiguous when B is stored column-major.
static __device__ __forceinline__ v16bf load_b_frag(const __bf16* p) {
  FragU f;
  f.u[0] = *reinterpret_cast<const uint4*>(p);
  f.u[1] = *reinterpret_cast<const uint4*>(p + 8);
  return f.v;
}

__global__ void prep_s(const float* __restrict__ ent, const int* __restrict__ sub,
                       __bf16* __restrict__ s_bf) {
  int i = blockIdx.x * blockDim.x + threadIdx.x;
  if (i >= BATCH * KP) return;
  int b = i / KP, k = i - b * KP;
  float v = (k < DE) ? ent[(size_t)sub[b] * DE + k] : 0.0f;
  s_bf[i] = (__bf16)v;
}

__global__ void zero_bf(__bf16* __restrict__ p, int n) {
  int i = blockIdx.x * blockDim.x + threadIdx.x;
  if (i < n) p[i] = (__bf16)0.0f;
}

__global__ void prep_ent(const float* __restrict__ ent, __bf16* __restrict__ ent_bf) {
  size_t i = (size_t)blockIdx.x * blockDim.x + threadIdx.x;
  if (i >= (size_t)NE * KP) return;
  size_t n = i / KP;
  int k = (int)(i - n * KP);
  float v = (k < DE) ? ent[n * DE + k] : 0.0f;
  ent_bf[i] = (__bf16)v;
}

// core[e,r,o] (f32, e-major) -> coreT[r][o][e] (bf16, e contiguous), zero pad.
// grid (DR, OPAD/16, KP/16), block (16,16); LDS tile transpose for coalescing.
__global__ void prep_coreT(const float* __restrict__ core, __bf16* __restrict__ coreT) {
  __shared__ float t[16][17];
  const int r  = blockIdx.x;
  const int o0 = blockIdx.y * 16;
  const int e0 = blockIdx.z * 16;
  const int tx = threadIdx.x, ty = threadIdx.y;
  int e = e0 + ty, o = o0 + tx;
  float v = 0.0f;
  if (e < DE && o < DE) v = core[(size_t)e * (DR * DE) + (size_t)r * DE + o];
  t[ty][tx] = v;
  __syncthreads();
  int oo = o0 + ty, ee = e0 + tx;
  coreT[((size_t)r * OPAD + oo) * KP + ee] = (__bf16)t[tx][ty];
}

// One wave per (16-b, 16-o) tile. For each r: 7 bf16 WMMAs (K=224) compute
// tmp1[16b,16o]; VALU epilogue accumulates r_vec[b,r]*tmp1 into tmp2.
// coreT is L2-resident (~18.6MB << 192MB), so the r-stream hits L2 not HBM.
__global__ void gemm1_contract(const __bf16* __restrict__ s_bf,
                               const __bf16* __restrict__ coreT,
                               const float* __restrict__ rel_emb,
                               const int*   __restrict__ rel_idx,
                               __bf16* __restrict__ tmp2b) {
  const int lane = threadIdx.x;
  const int half = lane >> 4;
  const int l15  = lane & 15;
  const int b0 = blockIdx.x * 16;
  const int o0 = blockIdx.y * 16;

  __shared__ float lds_r[16 * DR];
  for (int i = lane; i < 16 * DR; i += 32) {   // uniform trip count: EXEC stays full
    int m = i / DR, rr = i - m * DR;
    lds_r[i] = rel_emb[(size_t)rel_idx[b0 + m] * DR + rr];
  }
  __syncthreads();

  v16bf afrag[KP / 32];
#pragma unroll
  for (int kk = 0; kk < KP / 32; ++kk)
    afrag[kk] = load_a_frag(s_bf + (size_t)(b0 + l15) * KP + kk * 32 + half * 16);

  v8f outacc = {};
  const __bf16* bcol = coreT + (size_t)(o0 + l15) * KP + half * 16;
#pragma unroll 2
  for (int r = 0; r < DR; ++r) {
    const __bf16* bp = bcol + (size_t)r * OPAD * KP;
    v8f acc = {};
#pragma unroll
    for (int kk = 0; kk < KP / 32; ++kk) {
      v16bf bf = load_b_frag(bp + kk * 32);
      acc = __builtin_amdgcn_wmma_f32_16x16x32_bf16(false, afrag[kk], false, bf,
                                                    (short)0, acc, false, false);
    }
    // C/D layout: VGPR j <-> row (j + 8*half), col l15. Broadcast LDS reads.
#pragma unroll
    for (int j = 0; j < 8; ++j)
      outacc[j] += lds_r[(j + 8 * half) * DR + r] * acc[j];
  }

  const int o = o0 + l15;
  if (o < DE) {
#pragma unroll
    for (int j = 0; j < 8; ++j)
      tmp2b[(size_t)(b0 + j + 8 * half) * KP + o] = (__bf16)outacc[j];
  }
}

// One wave computes a 64x16 tile of logits: 4 stacked M sub-tiles reuse each
// B fragment (entity column), cutting entity re-reads to 16x (L2-resident).
// Sigmoid fused into the store epilogue.
__global__ void gemm2_sigmoid(const __bf16* __restrict__ tmp2b,
                              const __bf16* __restrict__ ent_bf,
                              float* __restrict__ out) {
  const int lane = threadIdx.x;
  const int half = lane >> 4;
  const int l15  = lane & 15;
  const int m0 = blockIdx.x * 64;
  const int n0 = blockIdx.y * 16;

  v8f acc[4] = {};
#pragma unroll
  for (int kk = 0; kk < KP / 32; ++kk) {
    v16bf bf = load_b_frag(ent_bf + (size_t)(n0 + l15) * KP + kk * 32 + half * 16);
#pragma unroll
    for (int m = 0; m < 4; ++m) {
      v16bf af = load_a_frag(tmp2b + (size_t)(m0 + m * 16 + l15) * KP + kk * 32 + half * 16);
      acc[m] = __builtin_amdgcn_wmma_f32_16x16x32_bf16(false, af, false, bf,
                                                       (short)0, acc[m], false, false);
    }
  }

  const int col = n0 + l15;
#pragma unroll
  for (int m = 0; m < 4; ++m) {
#pragma unroll
    for (int j = 0; j < 8; ++j) {
      int row = m0 + m * 16 + j + 8 * half;
      float x = acc[m][j];
      out[(size_t)row * NE + col] = 1.0f / (1.0f + __expf(-x));
    }
  }
}

extern "C" void kernel_launch(void* const* d_in, const int* in_sizes, int n_in,
                              void* d_out, int out_size, void* d_ws, size_t ws_size,
                              hipStream_t stream) {
  const float* core  = (const float*)d_in[0];
  // d_in[1] = grad_mask: forward value is independent of it (core*m + core*(1-m) == core)
  const float* ent   = (const float*)d_in[2];
  const float* rel   = (const float*)d_in[3];
  const int* sub_idx = (const int*)d_in[4];
  const int* rel_idx = (const int*)d_in[5];
  float* out = (float*)d_out;

  // Workspace layout (all 256B-aligned): ~64.4 MB total
  char* ws = (char*)d_ws;
  constexpr size_t coreT_bytes = (size_t)DR * OPAD * KP * 2;  // 18,636,800
  constexpr size_t ent_bytes   = (size_t)NE * KP * 2;         // 44,800,000
  constexpr size_t s_bytes     = (size_t)BATCH * KP * 2;      //    458,752
  __bf16* coreT  = (__bf16*)(ws);
  __bf16* ent_bf = (__bf16*)(ws + coreT_bytes);
  __bf16* s_bf   = (__bf16*)(ws + coreT_bytes + ent_bytes);
  __bf16* tmp2b  = (__bf16*)(ws + coreT_bytes + ent_bytes + s_bytes);
  (void)in_sizes; (void)n_in; (void)out_size; (void)ws_size;

  prep_s<<<(BATCH * KP + 255) / 256, 256, 0, stream>>>(ent, sub_idx, s_bf);
  zero_bf<<<(BATCH * KP + 255) / 256, 256, 0, stream>>>(tmp2b, BATCH * KP);
  prep_ent<<<(unsigned)(((size_t)NE * KP + 255) / 256), 256, 0, stream>>>(ent, ent_bf);
  prep_coreT<<<dim3(DR, OPAD / 16, KP / 16), dim3(16, 16), 0, stream>>>(core, coreT);
  gemm1_contract<<<dim3(BATCH / 16, OPAD / 16), 32, 0, stream>>>(s_bf, coreT, rel, rel_idx, tmp2b);
  gemm2_sigmoid<<<dim3(BATCH / 64, NE / 16), 32, 0, stream>>>(tmp2b, ent_bf, out);
}
